// TinyMoELM_22471268892970
// MI455X (gfx1250) — matrix operand order repaired and checked by reading, not compiled, not run on hardware
//
#include <hip/hip_runtime.h>
#include <math.h>

// ---- problem constants (match reference) ----
#define TT  2048      // tokens = B*S = 2*1024
#define HH  768       // hidden
#define DFF 3072      // ffn dim = 4*H
#define EE  8         // experts
#define LL  2         // layers
#define LN_EPS 1e-5f

typedef __attribute__((ext_vector_type(16))) __bf16 v16bf;
typedef __attribute__((ext_vector_type(8)))  __bf16 v8bf;
typedef __attribute__((ext_vector_type(8)))  float  v8f;
typedef __attribute__((ext_vector_type(4)))  float  v4f;

// fp32 -> bf16, round to nearest even
static __device__ __forceinline__ __bf16 f2bf(float f) {
  unsigned u = __float_as_uint(f);
  u += 0x7FFFu + ((u >> 16) & 1u);
  unsigned short s = (unsigned short)(u >> 16);
  __bf16 b;
  __builtin_memcpy(&b, &s, 2);
  return b;
}
static __device__ __forceinline__ float bf2f(__bf16 b) {
  unsigned short s;
  __builtin_memcpy(&s, &b, 2);
  return __uint_as_float(((unsigned)s) << 16);
}

// ---------------------------------------------------------------------------
// init: zero l_aux, counts (in d_out) and probsum accumulator (in ws)
// ---------------------------------------------------------------------------
__global__ void moe_init_kernel(float* __restrict__ laux_counts,
                                float* __restrict__ probsum) {
  int i = threadIdx.x;
  if (i < 9) laux_counts[i] = 0.0f;   // [0]=l_aux, [1..8]=counts
  if (i < 8) probsum[i] = 0.0f;
}

// ---------------------------------------------------------------------------
// bulk fp32 -> bf16 conversion (8 elements/thread, 16B vector stores).
// Memory-bound; keeps all conversion VALU OUT of the GEMM hot loops.
// ---------------------------------------------------------------------------
__global__ void __launch_bounds__(256) cvt_bf16_kernel(
    const float* __restrict__ src, __bf16* __restrict__ dst, int n8) {
  int i = blockIdx.x * 256 + threadIdx.x;
  if (i >= n8) return;
  const v4f* s = (const v4f*)(src + (size_t)i * 8);
  v4f q0 = s[0], q1 = s[1];
  v8bf o;
#pragma unroll
  for (int j = 0; j < 4; ++j) { o[j] = f2bf(q0[j]); o[4 + j] = f2bf(q1[j]); }
  *(v8bf*)(dst + (size_t)i * 8) = o;
}

// ---------------------------------------------------------------------------
// embed gather: x[t, :] = embed[ids[t], :]  (residual stream lives in d_out)
// ---------------------------------------------------------------------------
__global__ void __launch_bounds__(256) embed_kernel(
    const int* __restrict__ ids, const float* __restrict__ embed,
    float* __restrict__ x) {
  int t = blockIdx.x;
  size_t src = (size_t)ids[t] * HH;
  size_t dst = (size_t)t * HH;
  for (int h = threadIdx.x; h < HH; h += 256)
    x[dst + h] = embed[src + h];
}

// ---------------------------------------------------------------------------
// LayerNorm over H=768 per token, emit bf16 activations for WMMA
// ---------------------------------------------------------------------------
__global__ void __launch_bounds__(256) ln_kernel(
    const float* __restrict__ x, const float* __restrict__ g,
    const float* __restrict__ b, __bf16* __restrict__ flat) {
  __shared__ float red[256];
  const int t = blockIdx.x, tid = threadIdx.x;
  const float* xr = x + (size_t)t * HH;
  float v0 = xr[tid], v1 = xr[tid + 256], v2 = xr[tid + 512];

  red[tid] = v0 + v1 + v2;
  __syncthreads();
  for (int s = 128; s > 0; s >>= 1) { if (tid < s) red[tid] += red[tid + s]; __syncthreads(); }
  float mean = red[0] * (1.0f / HH);
  __syncthreads();

  float d0 = v0 - mean, d1 = v1 - mean, d2 = v2 - mean;
  red[tid] = d0 * d0 + d1 * d1 + d2 * d2;
  __syncthreads();
  for (int s = 128; s > 0; s >>= 1) { if (tid < s) red[tid] += red[tid + s]; __syncthreads(); }
  float rstd = rsqrtf(red[0] * (1.0f / HH) + LN_EPS);

  __bf16* fr = flat + (size_t)t * HH;
  fr[tid]       = f2bf(d0 * rstd * g[tid]       + b[tid]);
  fr[tid + 256] = f2bf(d1 * rstd * g[tid + 256] + b[tid + 256]);
  fr[tid + 512] = f2bf(d2 * rstd * g[tid + 512] + b[tid + 512]);
}

// ---------------------------------------------------------------------------
// gate: logits -> full softmax (for l_aux), top-2 + softmax-over-top2 -> wbuf,
// counts via atomics. E=8 is tiny; one thread per token is fine.
// ---------------------------------------------------------------------------
__global__ void __launch_bounds__(256) gate_kernel(
    const __bf16* __restrict__ flat, const float* __restrict__ gw,
    const float* __restrict__ gb, float* __restrict__ wbuf,
    float* __restrict__ probsum, float* __restrict__ counts) {
  int t = blockIdx.x * blockDim.x + threadIdx.x;
  if (t >= TT) return;
  float logit[EE];
#pragma unroll
  for (int e = 0; e < EE; ++e) logit[e] = gb[e];
  const __bf16* fr = flat + (size_t)t * HH;
  for (int h = 0; h < HH; ++h) {
    float xv = bf2f(fr[h]);
#pragma unroll
    for (int e = 0; e < EE; ++e) logit[e] += xv * gw[e * HH + h];
  }
  // full softmax for l_aux accumulation
  float m = logit[0];
#pragma unroll
  for (int e = 1; e < EE; ++e) m = fmaxf(m, logit[e]);
  float p[EE], s = 0.0f;
#pragma unroll
  for (int e = 0; e < EE; ++e) { p[e] = expf(logit[e] - m); s += p[e]; }
  float rs = 1.0f / s;
#pragma unroll
  for (int e = 0; e < EE; ++e) atomicAdd(&probsum[e], p[e] * rs);
  // top-2 (first index wins ties, like jax.lax.top_k)
  int i1 = 0;
#pragma unroll
  for (int e = 1; e < EE; ++e) if (logit[e] > logit[i1]) i1 = e;
  int i2 = (i1 == 0) ? 1 : 0;
#pragma unroll
  for (int e = 0; e < EE; ++e) if (e != i1 && logit[e] > logit[i2]) i2 = e;
  // softmax over the two selected logits (v1 >= v2)
  float e2 = expf(logit[i2] - logit[i1]);
  float inv = 1.0f / (1.0f + e2);
  float* wr = wbuf + (size_t)t * EE;
#pragma unroll
  for (int e = 0; e < EE; ++e) wr[e] = 0.0f;
  wr[i1] = inv;
  wr[i2] = e2 * inv;
  atomicAdd(&counts[i1], 1.0f);
  atomicAdd(&counts[i2], 1.0f);
}

__global__ void gate_fin_kernel(float* __restrict__ probsum,
                                float* __restrict__ laux) {
  if (threadIdx.x == 0 && blockIdx.x == 0) {
    float s = 0.0f;
    for (int e = 0; e < EE; ++e) {
      float p = probsum[e] * (1.0f / TT);
      s += p * p;
      probsum[e] = 0.0f;   // reset for next layer
    }
    *laux += s * (float)EE;
  }
}

// ---------------------------------------------------------------------------
// WMMA fragment loads (pure contiguous 16B per-lane loads; no conversions):
//   A 16x32 (M,K): lane<16 -> row M=lane, K in {k0..k0+7, k0+16..k0+23}
//                  lane>=16 -> same row, K in {k0+8..k0+15, k0+24..k0+31}
//   B 32x16 (K,N): lane gives N=lane%16; K same as A (k0+8*half..+7, k0+16+8*half..+7)
// ---------------------------------------------------------------------------
static __device__ __forceinline__ v16bf load_afrag(const __bf16* __restrict__ arow,
                                                   int k0, int half) {
  v8bf lo = *(const v8bf*)(arow + k0 + half * 8);
  v8bf hi = *(const v8bf*)(arow + k0 + 16 + half * 8);
  return __builtin_shufflevector(lo, hi, 0, 1, 2, 3, 4, 5, 6, 7,
                                 8, 9, 10, 11, 12, 13, 14, 15);
}
static __device__ __forceinline__ v16bf load_bfrag(const __bf16* __restrict__ wrow,
                                                   int k0, int half) {
  v8bf lo = *(const v8bf*)(wrow + k0 + half * 8);
  v8bf hi = *(const v8bf*)(wrow + k0 + 16 + half * 8);
  return __builtin_shufflevector(lo, hi, 0, 1, 2, 3, 4, 5, 6, 7,
                                 8, 9, 10, 11, 12, 13, 14, 15);
}

// ---------------------------------------------------------------------------
// fc1: h1 = gelu_exact(flat @ W1^T + b1), bf16 out.  M=T, N=DF, K=H.
// Wave tile 32x64: 2 A-frags + 4 B-frags -> 8 WMMAs per K-step.
// Block = 8 waves -> 256x64 C tile.
// ---------------------------------------------------------------------------
__global__ void __launch_bounds__(256) fc1_kernel(
    const __bf16* __restrict__ Abf,   // [T,H] bf16
    const __bf16* __restrict__ W,     // [DF,H] bf16 (expert slice, staged)
    const float* __restrict__ bias,   // [DF]
    __bf16* __restrict__ H1) {        // [T,DF] bf16
  const int wave = threadIdx.x >> 5;
  const int lane = threadIdx.x & 31;
  const int half = lane >> 4;
  const int lm   = lane & 15;
  const int row0 = blockIdx.y * 256 + wave * 32;
  const int col0 = blockIdx.x * 64;

  v8f acc[2][4] = {};
  const __bf16* arow0 = Abf + (size_t)(row0 + lm) * HH;
  const __bf16* arow1 = Abf + (size_t)(row0 + 16 + lm) * HH;

  for (int k0 = 0; k0 < HH; k0 += 32) {
    v16bf a0 = load_afrag(arow0, k0, half);
    v16bf a1 = load_afrag(arow1, k0, half);
#pragma unroll
    for (int j = 0; j < 4; ++j) {
      v16bf bf = load_bfrag(W + (size_t)(col0 + j * 16 + lm) * HH, k0, half);
      acc[0][j] = __builtin_amdgcn_wmma_f32_16x16x32_bf16(
          false, a0, false, bf, (short)0, acc[0][j], false, false);
      acc[1][j] = __builtin_amdgcn_wmma_f32_16x16x32_bf16(
          false, a1, false, bf, (short)0, acc[1][j], false, false);
    }
  }

#pragma unroll
  for (int mjt = 0; mjt < 2; ++mjt) {
#pragma unroll
    for (int j = 0; j < 4; ++j) {
      const int col = col0 + j * 16 + lm;
      const float bz = bias[col];
#pragma unroll
      for (int i = 0; i < 8; ++i) {
        const int row = row0 + mjt * 16 + i + half * 8;
        float v = acc[mjt][j][i] + bz;
        v = 0.5f * v * (1.0f + erff(v * 0.70710678118654752f));  // exact GELU
        H1[(size_t)row * DFF + col] = f2bf(v);
      }
    }
  }
}

// ---------------------------------------------------------------------------
// fc2: x += w[:,e] * (h1 @ W2^T + b2).  M=T, N=H, K=DF. Same tiling.
// Accumulates into the residual buffer (d_out x-region); experts run
// sequentially on the stream so plain RMW is safe (one lane per element).
// ---------------------------------------------------------------------------
__global__ void __launch_bounds__(256) fc2_kernel(
    const __bf16* __restrict__ H1,    // [T,DF] bf16
    const __bf16* __restrict__ W,     // [H,DF] bf16 (expert slice, staged)
    const float* __restrict__ bias,   // [H]
    const float* __restrict__ wbuf,   // [T,E] routing weights
    int e,
    float* __restrict__ xout) {       // [T,H] residual (accumulated)
  const int wave = threadIdx.x >> 5;
  const int lane = threadIdx.x & 31;
  const int half = lane >> 4;
  const int lm   = lane & 15;
  const int row0 = blockIdx.y * 256 + wave * 32;
  const int col0 = blockIdx.x * 64;

  v8f acc[2][4] = {};
  const __bf16* arow0 = H1 + (size_t)(row0 + lm) * DFF;
  const __bf16* arow1 = H1 + (size_t)(row0 + 16 + lm) * DFF;

  for (int k0 = 0; k0 < DFF; k0 += 32) {
    v16bf a0 = load_afrag(arow0, k0, half);
    v16bf a1 = load_afrag(arow1, k0, half);
#pragma unroll
    for (int j = 0; j < 4; ++j) {
      v16bf bf = load_bfrag(W + (size_t)(col0 + j * 16 + lm) * DFF, k0, half);
      acc[0][j] = __builtin_amdgcn_wmma_f32_16x16x32_bf16(
          false, a0, false, bf, (short)0, acc[0][j], false, false);
      acc[1][j] = __builtin_amdgcn_wmma_f32_16x16x32_bf16(
          false, a1, false, bf, (short)0, acc[1][j], false, false);
    }
  }

  float wrow[2][8];
#pragma unroll
  for (int mjt = 0; mjt < 2; ++mjt)
#pragma unroll
    for (int i = 0; i < 8; ++i)
      wrow[mjt][i] = wbuf[(size_t)(row0 + mjt * 16 + i + half * 8) * EE + e];

#pragma unroll
  for (int mjt = 0; mjt < 2; ++mjt) {
#pragma unroll
    for (int j = 0; j < 4; ++j) {
      const int col = col0 + j * 16 + lm;
      const float bz = bias[col];
#pragma unroll
      for (int i = 0; i < 8; ++i) {
        const int row = row0 + mjt * 16 + i + half * 8;
        const size_t idx = (size_t)row * HH + col;
        xout[idx] += wrow[mjt][i] * (acc[mjt][j][i] + bz);
      }
    }
  }
}

// ---------------------------------------------------------------------------
// launcher
// ---------------------------------------------------------------------------
extern "C" void kernel_launch(void* const* d_in, const int* in_sizes, int n_in,
                              void* d_out, int out_size, void* d_ws, size_t ws_size,
                              hipStream_t stream) {
  (void)in_sizes; (void)n_in; (void)out_size; (void)ws_size;
  const int*   ids    = (const int*)d_in[0];
  const float* embed  = (const float*)d_in[1];
  const float* ln_g   = (const float*)d_in[2];
  const float* ln_b   = (const float*)d_in[3];
  const float* gate_w = (const float*)d_in[4];
  const float* gate_b = (const float*)d_in[5];
  const float* fc1_w  = (const float*)d_in[6];
  const float* fc1_b  = (const float*)d_in[7];
  const float* fc2_w  = (const float*)d_in[8];
  const float* fc2_b  = (const float*)d_in[9];

  float* out_x      = (float*)d_out;                 // [T,H] residual/output
  float* out_laux   = out_x + (size_t)TT * HH;       // scalar
  float* out_counts = out_laux + 1;                  // [E]

  char* base = (char*)d_ws;
  __bf16* flatb = (__bf16*)base;                     // [T,H]   bf16 (3.1 MB)
  size_t off = (size_t)TT * HH * sizeof(__bf16);
  __bf16* h1 = (__bf16*)(base + off);                // [T,DF]  bf16 (12.6 MB)
  off += (size_t)TT * DFF * sizeof(__bf16);
  __bf16* w1b = (__bf16*)(base + off);               // [DF,H]  bf16 (4.7 MB)
  off += (size_t)DFF * HH * sizeof(__bf16);
  __bf16* w2b = (__bf16*)(base + off);               // [H,DF]  bf16 (4.7 MB)
  off += (size_t)HH * DFF * sizeof(__bf16);
  float* wbuf = (float*)(base + off);                // [T,E] fp32
  off += (size_t)TT * EE * sizeof(float);
  float* probsum = (float*)(base + off);             // [E]

  const int WN8 = (DFF * HH) / 8;                    // elems/8 per weight slice

  moe_init_kernel<<<1, 32, 0, stream>>>(out_laux, probsum);
  embed_kernel<<<TT, 256, 0, stream>>>(ids, embed, out_x);

  for (int l = 0; l < LL; ++l) {
    ln_kernel<<<TT, 256, 0, stream>>>(out_x, ln_g + l * HH, ln_b + l * HH, flatb);
    gate_kernel<<<TT / 256, 256, 0, stream>>>(flatb, gate_w + (size_t)l * EE * HH,
                                              gate_b + l * EE, wbuf, probsum,
                                              out_counts);
    gate_fin_kernel<<<1, 1, 0, stream>>>(probsum, out_laux);

    for (int e = 0; e < EE; ++e) {
      const size_t xi = (size_t)(l * EE + e);
      // stage expert weights as bf16 (memory-bound; off the GEMM critical VALU path)
      cvt_bf16_kernel<<<(WN8 + 255) / 256, 256, 0, stream>>>(
          fc1_w + xi * DFF * HH, w1b, WN8);
      cvt_bf16_kernel<<<(WN8 + 255) / 256, 256, 0, stream>>>(
          fc2_w + xi * (size_t)HH * DFF, w2b, WN8);

      dim3 g1(DFF / 64, TT / 256);
      fc1_kernel<<<g1, 256, 0, stream>>>(flatb, w1b, fc1_b + xi * DFF, h1);
      dim3 g2(HH / 64, TT / 256);
      fc2_kernel<<<g2, 256, 0, stream>>>(h1, w2b, fc2_b + xi * HH, wbuf, e, out_x);
    }
  }
}
